// WindowAttention_36197984371058
// MI455X (gfx1250) — matrix operand order repaired and checked by reading, not compiled, not run on hardware
//
#include <hip/hip_runtime.h>
#include <hip/hip_bf16.h>

// Window attention, fused: one block (256 thr = 8 wave32) per window, one wave per head.
// All GEMMs via v_wmma_f32_16x16x32_f16. Weights pre-converted to f16 in d_ws.

#define NTOK   64
#define DIMC   256
#define NHEADS 8
#define HD     32
#define SCALEQ 0.17677669529663687f  // 32^-0.5

typedef __attribute__((ext_vector_type(16))) _Float16 v16h;
typedef __attribute__((ext_vector_type(8)))  _Float16 v8h;
typedef __attribute__((ext_vector_type(4)))  _Float16 v4h;
typedef __attribute__((ext_vector_type(8)))  float    v8f;

// LDS layout (bytes)
#define OFF_XO    0         // 64 x 264 halfs (x as f16, later per-head concat output) = 33792 B
#define OFF_MASK  33792     // 64 x 68 floats = 17408 B
#define OFF_BIAS  51200     // 1800 floats = 7200 B
#define OFF_HEAD  58400     // 8 heads x 7424 halfs (Q 64x40 | K 64x40 | Vt 32x72); P reuses Q/K
#define SMEM_BYTES (58400 + 8*7424*2)   // 177184

static __device__ __forceinline__ v8f zero8() {
  v8f z = {0.f,0.f,0.f,0.f,0.f,0.f,0.f,0.f};
  return z;
}

static __device__ __forceinline__ v8f wmma_f16(v16h a, v16h b, v8f c) {
  // D = A(16x32 f16) * B(32x16 f16) + C(16x16 f32)
  return __builtin_amdgcn_wmma_f32_16x16x32_f16(false, a, false, b, (short)0, c, false, false);
}

// A fragment (16x32 f16): lane r holds row (rowBase+r); K chunks split per half-wave:
// halves of v16h at k = kbase + hf*8 .. +7 and kbase + hf*8 + 16 .. +23
static __device__ __forceinline__ v16h load_fragA(const _Float16* base, int row, int stride,
                                                  int kbase, int hf) {
  const _Float16* p = base + row * stride + kbase + hf * 8;
  v8h lo = *(const v8h*)p;
  v8h hi = *(const v8h*)(p + 16);
  v16h o;
#pragma unroll
  for (int e = 0; e < 8; ++e) { o[e] = lo[e]; o[e + 8] = hi[e]; }
  return o;
}

// B fragment (32x16 f16) loaded from B^T stored row-major (rows = N, cols = K):
// lane r holds column (colBase+r); K at kbase + hf*8 .. +7 and kbase + hf*8 + 16 .. +23.
static __device__ __forceinline__ v16h load_fragB(const _Float16* base, int row, int stride,
                                                  int kbase, int hf) {
  const _Float16* p = base + row * stride + kbase + hf * 8;
  v8h lo = *(const v8h*)p;
  v8h hi = *(const v8h*)(p + 16);
  v16h o;
#pragma unroll
  for (int e = 0; e < 8; ++e) { o[e] = lo[e]; o[e + 8] = hi[e]; }
  return o;
}

__global__ void convert_weights_kernel(const float* __restrict__ w_qkv,
                                       const float* __restrict__ w_proj,
                                       _Float16* __restrict__ wqkv_h,
                                       _Float16* __restrict__ wproj_h) {
  int i = blockIdx.x * blockDim.x + threadIdx.x;   // grid covers 768*256 = 196608
  wqkv_h[i] = (_Float16)w_qkv[i];
  if (i < DIMC * DIMC) wproj_h[i] = (_Float16)w_proj[i];
}

__global__ __launch_bounds__(256)
void win_attn_kernel(const float* __restrict__ x, const float* __restrict__ mask,
                     const float* __restrict__ b_qkv, const float* __restrict__ b_proj,
                     const float* __restrict__ bias_table,
                     const _Float16* __restrict__ wqkv_h, const _Float16* __restrict__ wproj_h,
                     float* __restrict__ out) {
  extern __shared__ char smem[];
  _Float16* XO    = (_Float16*)(smem + OFF_XO);    // x f16 then per-head output, stride 264
  float*    maskS = (float*)(smem + OFF_MASK);     // stride 68
  float*    biasS = (float*)(smem + OFF_BIAS);     // flat 225*8

  const int t    = threadIdx.x;
  const int w    = t >> 5;        // wave id == head id
  const int lane = t & 31;
  const int r    = lane & 15;
  const int hf   = lane >> 4;
  const int b    = blockIdx.x;

  _Float16* Q  = (_Float16*)(smem + OFF_HEAD) + w * 7424;  // 64 x 40
  _Float16* K  = Q + 2560;                                  // 64 x 40
  _Float16* Vt = K + 2560;                                  // 32 x 72 (features x tokens)
  _Float16* P  = Q;                                         // probs 64 x 72, reuses Q/K

  // ---- stage x (f32 -> f16 in LDS, stride 264) ----
  const float* xg = x + (size_t)b * (NTOK * DIMC);
#pragma unroll
  for (int it = 0; it < 16; ++it) {
    int idx = it * 1024 + t * 4;
    float4 vv = *(const float4*)(xg + idx);
    int row = idx >> 8, col = idx & 255;
    v4h h4 = {(_Float16)vv.x, (_Float16)vv.y, (_Float16)vv.z, (_Float16)vv.w};
    *(v4h*)(XO + row * 264 + col) = h4;
  }
  // ---- stage mask (f32, stride 68) ----
  const float* mg = mask + (size_t)b * (NTOK * NTOK);
#pragma unroll
  for (int it = 0; it < 4; ++it) {
    int idx = it * 1024 + t * 4;
    float4 vv = *(const float4*)(mg + idx);
    int row = idx >> 6, col = idx & 63;
    *(float4*)(maskS + row * 68 + col) = vv;
  }
  // ---- stage bias table ----
  for (int idx = t; idx < 1800; idx += 256) biasS[idx] = bias_table[idx];
  __syncthreads();

  // ================= Phase 1: qkv = x @ w_qkv^T (per-head slices) =================
  for (int p = 0; p < 3; ++p) {   // 0=q, 1=k, 2=v
    v8f acc[4][2];
#pragma unroll
    for (int mi = 0; mi < 4; ++mi)
#pragma unroll
      for (int nj = 0; nj < 2; ++nj) acc[mi][nj] = zero8();

    for (int ks = 0; ks < 8; ++ks) {   // K = 256 in steps of 32
      v16h bf[2];
#pragma unroll
      for (int nj = 0; nj < 2; ++nj)
        bf[nj] = load_fragB(wqkv_h, p * 256 + w * 32 + nj * 16 + r, 256, ks * 32, hf);
#pragma unroll
      for (int mi = 0; mi < 4; ++mi) {
        v16h a = load_fragA(XO, mi * 16 + r, 264, ks * 32, hf);
#pragma unroll
        for (int nj = 0; nj < 2; ++nj) acc[mi][nj] = wmma_f16(a, bf[nj], acc[mi][nj]);
      }
    }
    float bq0 = b_qkv[p * 256 + w * 32 + r];
    float bq1 = b_qkv[p * 256 + w * 32 + 16 + r];
#pragma unroll
    for (int mi = 0; mi < 4; ++mi) {
#pragma unroll
      for (int nj = 0; nj < 2; ++nj) {
        float bq = (nj == 0) ? bq0 : bq1;
        int d = nj * 16 + r;
        if (p == 2) {  // v: store transposed, pack 8 consecutive tokens
          v8h pk;
#pragma unroll
          for (int i = 0; i < 8; ++i) pk[i] = (_Float16)(acc[mi][nj][i] + bq);
          *(v8h*)(Vt + d * 72 + mi * 16 + hf * 8) = pk;
        } else {
          _Float16* dst = (p == 0) ? Q : K;
          float sc = (p == 0) ? SCALEQ : 1.0f;
#pragma unroll
          for (int i = 0; i < 8; ++i) {
            int m = mi * 16 + hf * 8 + i;
            dst[m * 40 + d] = (_Float16)((acc[mi][nj][i] + bq) * sc);
          }
        }
      }
    }
  }
  __syncthreads();   // all waves done reading XO (x); XO becomes output staging

  // ================= Phase 2: S = q @ k^T (+bias+mask), softmax =================
  v8f s[4][4];
  {
    v16h aq[4];
#pragma unroll
    for (int mi = 0; mi < 4; ++mi) aq[mi] = load_fragA(Q, mi * 16 + r, 40, 0, hf);
#pragma unroll
    for (int nj = 0; nj < 4; ++nj) {
      v16h bk = load_fragB(K, nj * 16 + r, 40, 0, hf);
#pragma unroll
      for (int mi = 0; mi < 4; ++mi) s[mi][nj] = wmma_f16(aq[mi], bk, zero8());
    }
  }
#pragma unroll
  for (int mi = 0; mi < 4; ++mi) {
#pragma unroll
    for (int i = 0; i < 8; ++i) {
      int m = mi * 16 + hf * 8 + i;
      int ym = m >> 3, xm = m & 7;
      float sv[4];
      float rowmax = -3.0e38f;
#pragma unroll
      for (int nj = 0; nj < 4; ++nj) {
        int n = nj * 16 + r;
        int yn = n >> 3, xn = n & 7;
        int rel = (ym - yn + 7) * 15 + (xm - xn + 7);
        float v = s[mi][nj][i] + biasS[rel * 8 + w] + maskS[m * 68 + n];
        sv[nj] = v;
        rowmax = fmaxf(rowmax, v);
      }
#pragma unroll
      for (int d = 1; d < 16; d <<= 1) rowmax = fmaxf(rowmax, __shfl_xor(rowmax, d, 32));
      float rsum = 0.f;
#pragma unroll
      for (int nj = 0; nj < 4; ++nj) { sv[nj] = __expf(sv[nj] - rowmax); rsum += sv[nj]; }
#pragma unroll
      for (int d = 1; d < 16; d <<= 1) rsum += __shfl_xor(rsum, d, 32);
      float inv = 1.0f / rsum;
#pragma unroll
      for (int nj = 0; nj < 4; ++nj)
        P[m * 72 + nj * 16 + r] = (_Float16)(sv[nj] * inv);   // P reuses Q/K area (reads done)
    }
  }

  // ================= Phase 3: O_h = P @ V =================
  v8f o2[4][2];
#pragma unroll
  for (int mi = 0; mi < 4; ++mi)
#pragma unroll
    for (int nj = 0; nj < 2; ++nj) o2[mi][nj] = zero8();
#pragma unroll
  for (int mi = 0; mi < 4; ++mi) {
#pragma unroll
    for (int ks = 0; ks < 2; ++ks) {   // K = 64 tokens, steps of 32
      v16h a = load_fragA(P, mi * 16 + r, 72, ks * 32, hf);
#pragma unroll
      for (int nj = 0; nj < 2; ++nj) {
        v16h bv = load_fragB(Vt, nj * 16 + r, 72, ks * 32, hf);
        o2[mi][nj] = wmma_f16(a, bv, o2[mi][nj]);
      }
    }
  }
#pragma unroll
  for (int mi = 0; mi < 4; ++mi)
#pragma unroll
    for (int nj = 0; nj < 2; ++nj)
#pragma unroll
      for (int i = 0; i < 8; ++i) {
        int m = mi * 16 + hf * 8 + i;
        XO[m * 264 + w * 32 + nj * 16 + r] = (_Float16)o2[mi][nj][i];
      }
  __syncthreads();   // all heads' slices written

  // ================= Phase 4: out = O @ w_proj^T + b_proj =================
  v8f acc2[4][2];
#pragma unroll
  for (int mi = 0; mi < 4; ++mi)
#pragma unroll
    for (int nj = 0; nj < 2; ++nj) acc2[mi][nj] = zero8();
  for (int ks = 0; ks < 8; ++ks) {
    v16h bf[2];
#pragma unroll
    for (int nj = 0; nj < 2; ++nj)
      bf[nj] = load_fragB(wproj_h, w * 32 + nj * 16 + r, 256, ks * 32, hf);
#pragma unroll
    for (int mi = 0; mi < 4; ++mi) {
      v16h a = load_fragA(XO, mi * 16 + r, 264, ks * 32, hf);
#pragma unroll
      for (int nj = 0; nj < 2; ++nj) acc2[mi][nj] = wmma_f16(a, bf[nj], acc2[mi][nj]);
    }
  }
  float bp0 = b_proj[w * 32 + r];
  float bp1 = b_proj[w * 32 + 16 + r];
  float* og = out + (size_t)b * (NTOK * DIMC);
#pragma unroll
  for (int mi = 0; mi < 4; ++mi)
#pragma unroll
    for (int nj = 0; nj < 2; ++nj)
#pragma unroll
      for (int i = 0; i < 8; ++i) {
        int m = mi * 16 + hf * 8 + i;
        og[m * 256 + w * 32 + nj * 16 + r] = acc2[mi][nj][i] + ((nj == 0) ? bp0 : bp1);
      }
}

extern "C" void kernel_launch(void* const* d_in, const int* in_sizes, int n_in,
                              void* d_out, int out_size, void* d_ws, size_t ws_size,
                              hipStream_t stream) {
  (void)in_sizes; (void)n_in; (void)out_size; (void)ws_size;
  const float* x          = (const float*)d_in[0];
  const float* mask       = (const float*)d_in[1];
  const float* w_qkv      = (const float*)d_in[2];
  const float* b_qkv      = (const float*)d_in[3];
  const float* w_proj     = (const float*)d_in[4];
  const float* b_proj     = (const float*)d_in[5];
  const float* bias_table = (const float*)d_in[6];
  float* out = (float*)d_out;

  _Float16* wqkv_h  = (_Float16*)d_ws;            // 768*256 halfs
  _Float16* wproj_h = wqkv_h + 768 * 256;         // 256*256 halfs

  convert_weights_kernel<<<768, 256, 0, stream>>>(w_qkv, w_proj, wqkv_h, wproj_h);

  (void)hipFuncSetAttribute((const void*)win_attn_kernel,
                            hipFuncAttributeMaxDynamicSharedMemorySize, SMEM_BYTES);
  win_attn_kernel<<<2048, 256, SMEM_BYTES, stream>>>(x, mask, b_qkv, b_proj, bias_table,
                                                     wqkv_h, wproj_h, out);
}
